// BoronNMRNet_V3_62002147885455
// MI455X (gfx1250) — compile-verified, matrix operand order checked by reading, not run on hardware
//
#include <hip/hip_runtime.h>
#include <hip/hip_bf16.h>
#include <stdint.h>

// ---------------------------------------------------------------------------
// CDNA5 (gfx1250) WMMA types
// ---------------------------------------------------------------------------
typedef __attribute__((ext_vector_type(16))) __bf16 v16bf;
typedef __attribute__((ext_vector_type(8)))  float  v8f;

#define HIDDIM 256
#define NHEADS 4
#define HEADD  64

static inline int cdiv_l(long long a, long long b) { return (int)((a + b - 1) / b); }

// pack two f32 into two bf16 (RNE) in one u32
__device__ __forceinline__ uint32_t packbf2(float a, float b) {
    unsigned ua = __float_as_uint(a);
    unsigned ub = __float_as_uint(b);
    ua = ua + 0x7FFFu + ((ua >> 16) & 1u);
    ub = ub + 0x7FFFu + ((ub >> 16) & 1u);
    return (ua >> 16) | (ub & 0xFFFF0000u);
}

// ---------------------------------------------------------------------------
// Tensor Data Mover staging (gfx1250): stage a 2D tile [tileRows x 32 bf16]
// from a row-major bf16 matrix into LDS with a 16B pad after each 64B row,
// giving an LDS row stride of 80B (20 u32) = bank-conflict-free WMMA reads.
// D# layout per CDNA5 ISA 8.3-8.5 (group0: count/lds/global/type,
// group1: data_size, pad, tensor dims, tile dims, stride).
// ---------------------------------------------------------------------------
#if defined(__gfx1250__) && __has_builtin(__builtin_amdgcn_tensor_load_to_lds)
#define GEMM_USE_TDM 1
typedef __attribute__((ext_vector_type(4))) unsigned int u32x4_t;
typedef __attribute__((ext_vector_type(8))) int i32x8_t;
typedef __attribute__((ext_vector_type(4))) int i32x4_t;

__device__ __forceinline__ void tdm_tile_load(uint32_t ldsByteOff, const void* gsrc,
                                              int tileRows, int tensorRows,
                                              int strideHalves)
{
    uint64_t ga = (uint64_t)(uintptr_t)gsrc;
    u32x4_t g0;
    g0[0] = 1u;                                     // count=1 (valid D#), user mode
    g0[1] = ldsByteOff;                             // lds_addr (bytes)
    g0[2] = (uint32_t)ga;                           // global_addr[31:0]
    g0[3] = (uint32_t)((ga >> 32) & 0x01FFFFFFu)    // global_addr[56:32]
          | (2u << 30);                             // type = 2 ("image")
    uint32_t trLo = (uint32_t)tensorRows & 0xFFFFu;
    uint32_t trHi = ((uint32_t)tensorRows >> 16) & 0xFFFFu;
    i32x8_t g1;
    g1[0] = (int)((1u << 16)      // data_size = 2 bytes
                | (1u << 20)      // pad_enable
                | (3u << 22)      // pad_interval: 16 DWORDs (64B)
                | (3u << 25));    // pad_amount: 4 DWORDs (16B)
    g1[1] = (int)(32u << 16);     // tensor_dim0[15:0] = 32 halves
    g1[2] = (int)(trLo << 16);    // tensor_dim0[31:16]=0 | tensor_dim1[15:0]
    g1[3] = (int)(trHi | (32u << 16));  // tensor_dim1[31:16] | tile_dim0 = 32
    g1[4] = (int)(uint32_t)tileRows;    // tile_dim1 | tile_dim2 = 0
    g1[5] = (int)(uint32_t)strideHalves;// tensor_dim0_stride[31:0] (data_size units)
    g1[6] = 0;                          // stride[47:32] | tensor_dim1_stride lo
    g1[7] = 0;
    i32x4_t z4 = {0, 0, 0, 0};                   // groups 2/3 unused (2D tensor)
    i32x8_t z8 = {0, 0, 0, 0, 0, 0, 0, 0};       // 6-arg toolchain extra group
    __builtin_amdgcn_tensor_load_to_lds(g0, g1, z4, z4, z8, 0);
}
#endif

// ---------------------------------------------------------------------------
// GEMM: C[M,N] = A[M,Kpad] @ Bt[N,Kpad]^T (+bias[N]) (+addend[M,N]) (relu)
// A, Bt are pre-converted bf16 (pair-packed u32 rows, K-contiguous).
// Block tile 128x64, 256 threads (8 waves), K-step 32, double-buffered LDS,
// staged via TDM tensor_load_to_lds (fallback: vectorized copies).
// Requires N % 64 == 0, Kpad % 32 == 0.
// ---------------------------------------------------------------------------
#define AROWU 20   // LDS row stride in u32: 16 data + 4 pad (80B)

__global__ __launch_bounds__(256)
void gemm_wmma_bf16(const uint32_t* __restrict__ A, const uint32_t* __restrict__ Bt,
                    const float* __restrict__ bias, const float* __restrict__ addend,
                    float* __restrict__ C, int M, int N, int Kpad, int relu)
{
    __shared__ uint32_t sA[2][128 * AROWU];
    __shared__ uint32_t sB[2][64 * AROWU];

    const int tid  = threadIdx.x;
    const int lane = tid & 31;
    const int wid  = tid >> 5;
    const int rowBase = blockIdx.x * 128;
    const int colBase = blockIdx.y * 64;
    const int kw = Kpad >> 1;   // u32 per row

    v8f acc[4];
#pragma unroll
    for (int c = 0; c < 4; ++c)
#pragma unroll
        for (int j = 0; j < 8; ++j) acc[c][j] = 0.0f;

    const int r   = lane & 15;        // fragment row (A) / col (B^T)
    const int ofs = (lane >> 4) * 4;  // u32 offset: K-halves 0..7 vs 8..15

    // Load A fragment + ALL FOUR B fragments into distinct registers first,
    // then issue the 4 WMMAs back-to-back (single dscnt wait, XDL pipelining).
    auto computeStep = [&](int pp) {
        v16bf afrag;
        v16bf bfrag[4];
        {
            uint32_t* au = reinterpret_cast<uint32_t*>(&afrag);
            const uint32_t* ap = &sA[pp][(16 * wid + r) * AROWU];
#pragma unroll
            for (int i = 0; i < 4; ++i) {
                au[i]     = ap[ofs + i];      // K = kb+0..7
                au[4 + i] = ap[8 + ofs + i];  // K = kb+16..23
            }
        }
#pragma unroll
        for (int c = 0; c < 4; ++c) {
            uint32_t* bu = reinterpret_cast<uint32_t*>(&bfrag[c]);
            const uint32_t* bp = &sB[pp][(16 * c + r) * AROWU];
#pragma unroll
            for (int i = 0; i < 4; ++i) {
                bu[i]     = bp[ofs + i];
                bu[4 + i] = bp[8 + ofs + i];
            }
        }
#pragma unroll
        for (int c = 0; c < 4; ++c) {
            acc[c] = __builtin_amdgcn_wmma_f32_16x16x32_bf16(
                false, afrag, false, bfrag[c], (short)0, acc[c], false, false);
        }
    };

#ifdef GEMM_USE_TDM
    // ---- TDM staging: one wave issues async tile DMAs, TENSORcnt + barrier ----
    auto issueTiles = [&](int k0, int pp) {
        tdm_tile_load((uint32_t)(uintptr_t)&sA[pp][0],
                      A + (size_t)rowBase * kw + (k0 >> 1),
                      128, M - rowBase, Kpad);
        tdm_tile_load((uint32_t)(uintptr_t)&sB[pp][0],
                      Bt + (size_t)colBase * kw + (k0 >> 1),
                      64, 64, Kpad);
    };
    if (wid == 0) issueTiles(0, 0);
    int p = 0;
    for (int k0 = 0; k0 < Kpad; k0 += 32) {
        if (wid == 0) __builtin_amdgcn_s_wait_tensorcnt(0);
        __syncthreads();
        if (wid == 0 && k0 + 32 < Kpad) issueTiles(k0 + 32, p ^ 1);
        computeStep(p);
        p ^= 1;
    }
#else
    // ---- fallback staging: vectorized global->reg->LDS, double-buffered ----
    const int arow = tid >> 1, aoff = (tid & 1) * 8;   // A: row, u32 offset 0/8
    const int brow = tid >> 2, boff = (tid & 3) * 2;   // B: row, u32 offset 0..6
    uint4 aV0, aV1;
    uint2 bV;
    auto loadTiles = [&](int k0) {
        const uint32_t* ga = A + (size_t)(rowBase + arow) * kw + (k0 >> 1) + aoff;
        aV0 = *(const uint4*)(ga);
        aV1 = *(const uint4*)(ga + 4);
        const uint32_t* gb = Bt + (size_t)(colBase + brow) * kw + (k0 >> 1) + boff;
        bV = *(const uint2*)(gb);
    };
    auto storeTiles = [&](int pp) {
        *(uint4*)&sA[pp][arow * AROWU + aoff]     = aV0;
        *(uint4*)&sA[pp][arow * AROWU + aoff + 4] = aV1;
        *(uint2*)&sB[pp][brow * AROWU + boff]     = bV;
    };
    loadTiles(0);
    int p = 0;
    for (int k0 = 0; k0 < Kpad; k0 += 32) {
        storeTiles(p);
        if (k0 + 32 < Kpad) loadTiles(k0 + 32);
        __syncthreads();
        computeStep(p);
        p ^= 1;
    }
#endif

    // ---- epilogue: C VGPR j -> (m = j + 8*(lane>>4), n = lane&15) ----
    const int mo = (lane >> 4) * 8;
    if (rowBase + 128 <= M) {
#pragma unroll
        for (int c = 0; c < 4; ++c) {
            int gcol = colBase + 16 * c + r;
            float bv = bias ? bias[gcol] : 0.0f;
#pragma unroll
            for (int j = 0; j < 8; ++j) {
                int grow = rowBase + 16 * wid + mo + j;
                size_t idx = (size_t)grow * N + gcol;
                float v = acc[c][j] + bv;
                if (addend) v += addend[idx];
                if (relu)   v = v > 0.0f ? v : 0.0f;
                C[idx] = v;
            }
        }
    } else {
#pragma unroll
        for (int c = 0; c < 4; ++c) {
            int gcol = colBase + 16 * c + r;
            float bv = bias ? bias[gcol] : 0.0f;
#pragma unroll
            for (int j = 0; j < 8; ++j) {
                int grow = rowBase + 16 * wid + mo + j;
                if (grow < M) {
                    size_t idx = (size_t)grow * N + gcol;
                    float v = acc[c][j] + bv;
                    if (addend) v += addend[idx];
                    if (relu)   v = v > 0.0f ? v : 0.0f;
                    C[idx] = v;
                }
            }
        }
    }
}

// ---------------------------------------------------------------------------
// Conversion kernels (f32 -> pair-packed bf16)
// ---------------------------------------------------------------------------
// X[rows][K] f32 -> Y[rows][Kpad/2] u32 (zero-pad K..Kpad); K even.
__global__ void cvt_pack_kernel(const float* __restrict__ X, uint32_t* __restrict__ Y,
                                int total, int K, int Kpad) {
    int t = blockIdx.x * blockDim.x + threadIdx.x;
    if (t >= total) return;
    int kwp = Kpad >> 1;
    int row = t / kwp;
    int k = (t - row * kwp) * 2;
    uint32_t y = 0u;
    if (k < K) {
        float2 ab = *(const float2*)(X + (size_t)row * K + k);
        y = packbf2(ab.x, ab.y);
    }
    Y[t] = y;
}
// W[K][N] f32 -> Wt[N][Kpad/2] u32 (transpose + pack); K even.
__global__ void cvtT_pack_kernel(const float* __restrict__ W, uint32_t* __restrict__ Wt,
                                 int total, int K, int N, int Kpad) {
    int t = blockIdx.x * blockDim.x + threadIdx.x;
    if (t >= total) return;
    int kwp = Kpad >> 1;
    int n = t / kwp;
    int k = (t - n * kwp) * 2;
    float a = (k < K)     ? W[(size_t)k * N + n]       : 0.0f;
    float b = (k + 1 < K) ? W[(size_t)(k + 1) * N + n] : 0.0f;
    Wt[t] = packbf2(a, b);
}

// ---------------------------------------------------------------------------
// Elementwise / scatter / reduction kernels
// ---------------------------------------------------------------------------
__global__ void zero_f32_kernel(float* p, long long n) {
    long long t = (long long)blockIdx.x * blockDim.x + threadIdx.x;
    if (t < n) p[t] = 0.0f;
}
__global__ void zero_u32_kernel(unsigned* p, long long n) {
    long long t = (long long)blockIdx.x * blockDim.x + threadIdx.x;
    if (t < n) p[t] = 0u;
}
__global__ void copy_f32_kernel(const float* __restrict__ s, float* __restrict__ d, long long n) {
    long long t = (long long)blockIdx.x * blockDim.x + threadIdx.x;
    if (t < n) d[t] = s[t];
}
__global__ void bcast_row_kernel(const float* __restrict__ row, float* __restrict__ out, long long total) {
    long long t = (long long)blockIdx.x * blockDim.x + threadIdx.x;
    if (t < total) out[t] = row[t & (HIDDIM - 1)];
}
// hh_bf16[n][pair] = pack(h[n] + vn[batch[n]])  (direct bf16 emission)
__global__ void add_vn_cvt_kernel(const float* __restrict__ h, const float* __restrict__ vn,
                                  const int* __restrict__ batch, uint32_t* __restrict__ hb,
                                  int total /* = N*128 */) {
    int t = blockIdx.x * blockDim.x + threadIdx.x;
    if (t >= total) return;
    int n = t >> 7;
    int k = (t & 127) * 2;
    float2 hv = *(const float2*)(h + (size_t)n * HIDDIM + k);
    float2 vv = *(const float2*)(vn + (size_t)batch[n] * HIDDIM + k);
    hb[t] = packbf2(hv.x + vv.x, hv.y + vv.y);
}

// ordered-uint encoding for float atomic max
__device__ __forceinline__ unsigned ordenc(float f) {
    unsigned b = __float_as_uint(f);
    return (b & 0x80000000u) ? ~b : (b | 0x80000000u);
}
__device__ __forceinline__ float orddec(unsigned u) {
    if (u == 0u) return 0.0f;  // empty segment: reference maps -inf -> 0
    return (u & 0x80000000u) ? __uint_as_float(u ^ 0x80000000u) : __uint_as_float(~u);
}

// pass 1: alpha[e,h] = (q[dst] . (k[src]+ee[e])) * 0.125 ; segment max
__global__ void attn_alpha_kernel(const float* __restrict__ Q, const float* __restrict__ Kx,
                                  const float* __restrict__ EE,
                                  const int* __restrict__ src, const int* __restrict__ dst,
                                  unsigned* __restrict__ m_enc, float* __restrict__ alpha,
                                  long long total) {
    long long t = (long long)blockIdx.x * blockDim.x + threadIdx.x;
    if (t >= total) return;
    int e = (int)(t >> 2);
    int h = (int)(t & 3);
    int s = src[e], d = dst[e];
    const float* qp = Q  + (size_t)d * HIDDIM + h * HEADD;
    const float* kp = Kx + (size_t)s * HIDDIM + h * HEADD;
    const float* ep = EE + (size_t)e * HIDDIM + h * HEADD;
    float acc = 0.0f;
#pragma unroll
    for (int i = 0; i < HEADD; i += 4) {
        float4 q4 = *(const float4*)(qp + i);
        float4 k4 = *(const float4*)(kp + i);
        float4 e4 = *(const float4*)(ep + i);
        acc += q4.x * (k4.x + e4.x) + q4.y * (k4.y + e4.y)
             + q4.z * (k4.z + e4.z) + q4.w * (k4.w + e4.w);
    }
    acc *= 0.125f;
    alpha[t] = acc;
    atomicMax(&m_enc[(size_t)d * NHEADS + h], ordenc(acc));
}

// pass 2: ea = exp(alpha - m[dst]); s[dst] += ea  (in-place over alpha)
__global__ void attn_expsum_kernel(const int* __restrict__ dst, const unsigned* __restrict__ m_enc,
                                   float* __restrict__ alpha, float* __restrict__ ssum,
                                   long long total) {
    long long t = (long long)blockIdx.x * blockDim.x + threadIdx.x;
    if (t >= total) return;
    int e = (int)(t >> 2);
    int h = (int)(t & 3);
    int d = dst[e];
    float m  = orddec(m_enc[(size_t)d * NHEADS + h]);
    float ea = __expf(alpha[t] - m);
    alpha[t] = ea;
    atomicAdd(&ssum[(size_t)d * NHEADS + h], ea);
}

// pass 3: O[dst] += (v[src]+ee) * ea/(s[dst]+1e-16)
__global__ void attn_scatter_kernel(const float* __restrict__ V, const float* __restrict__ EE,
                                    const float* __restrict__ ea, const float* __restrict__ ssum,
                                    const int* __restrict__ src, const int* __restrict__ dst,
                                    float* __restrict__ O, long long total) {
    long long t = (long long)blockIdx.x * blockDim.x + threadIdx.x;
    if (t >= total) return;
    int e = (int)(t >> 6);
    int c = (int)(t & 63) * 4;
    int h = c >> 6;
    int s = src[e], d = dst[e];
    float a = ea[(size_t)e * NHEADS + h] / (ssum[(size_t)d * NHEADS + h] + 1e-16f);
    float4 v4 = *(const float4*)(V  + (size_t)s * HIDDIM + c);
    float4 e4 = *(const float4*)(EE + (size_t)e * HIDDIM + c);
    float* op = O + (size_t)d * HIDDIM + c;
    atomicAdd(op + 0, (v4.x + e4.x) * a);
    atomicAdd(op + 1, (v4.y + e4.y) * a);
    atomicAdd(op + 2, (v4.z + e4.z) * a);
    atomicAdd(op + 3, (v4.w + e4.w) * a);
}

// batchnorm column sums (thread = channel; coalesced row stream + prefetch)
__global__ __launch_bounds__(256)
void colsum_kernel(const float* __restrict__ X, float* __restrict__ sum,
                   float* __restrict__ sumsq, int M) {
    int ch = threadIdx.x;
    int r0 = blockIdx.x * 128;
    int r1 = r0 + 128 < M ? r0 + 128 : M;
    float s = 0.0f, q = 0.0f;
    for (int r = r0; r < r1; ++r) {
        const float* pr = X + (size_t)r * HIDDIM + ch;
        __builtin_prefetch(pr + 8 * HIDDIM, 0, 0);   // global_prefetch_b8
        float v = *pr;
        s += v; q += v * v;
    }
    atomicAdd(&sum[ch], s);
    atomicAdd(&sumsq[ch], q);
}
__global__ void bnstats_finalize_kernel(const float* __restrict__ sum, const float* __restrict__ sumsq,
                                        float* __restrict__ mean, float* __restrict__ rstd, int M) {
    int ch = threadIdx.x;
    float inv = 1.0f / (float)M;
    float mu  = sum[ch] * inv;
    float var = sumsq[ch] * inv - mu * mu;
    mean[ch] = mu;
    rstd[ch] = rsqrtf(var + 1e-5f);
}
__global__ void bn_relu_res_kernel(const float* __restrict__ X, const float* __restrict__ mean,
                                   const float* __restrict__ rstd, const float* __restrict__ g,
                                   const float* __restrict__ b, const float* __restrict__ resid,
                                   float* __restrict__ OUT, long long total) {
    long long t = (long long)blockIdx.x * blockDim.x + threadIdx.x;
    if (t >= total) return;
    int ch = (int)(t & 255);
    float v = (X[t] - mean[ch]) * rstd[ch] * g[ch] + b[ch];
    v = v > 0.0f ? v : 0.0f;
    OUT[t] = resid ? v + resid[t] : v;
}

__global__ void pool_scatter_kernel(const float* __restrict__ h, const int* __restrict__ batch,
                                    float* __restrict__ pooled, long long total) {
    long long t = (long long)blockIdx.x * blockDim.x + threadIdx.x;
    if (t >= total) return;
    int n = (int)(t >> 6);
    int c = (int)(t & 63) * 4;
    int g = batch[n];
    float4 v = *(const float4*)(h + (size_t)n * HIDDIM + c);
    float* pp = pooled + (size_t)g * HIDDIM + c;
    atomicAdd(pp + 0, v.x);
    atomicAdd(pp + 1, v.y);
    atomicAdd(pp + 2, v.z);
    atomicAdd(pp + 3, v.w);
}

// gather boron rows + solvent embedding -> comb_bf16[NB][160] (pair-packed)
__global__ void gather_comb_cvt_kernel(const float* __restrict__ h, const float* __restrict__ solvent_emb,
                                       const int* __restrict__ solvent_ids, const int* __restrict__ b_index,
                                       const int* __restrict__ batch, uint32_t* __restrict__ comb,
                                       int total /* NB*160 */) {
    int t = blockIdx.x * blockDim.x + threadIdx.x;
    if (t >= total) return;
    int i = t / 160;
    int c = (t - i * 160) * 2;
    int node = b_index[i];
    float2 ab;
    if (c < HIDDIM) {
        ab = *(const float2*)(h + (size_t)node * HIDDIM + c);
    } else {
        int g   = batch[node];
        int sid = solvent_ids[g];
        ab = *(const float2*)(solvent_emb + (size_t)sid * 64 + (c - HIDDIM));
    }
    comb[t] = packbf2(ab.x, ab.y);
}

__global__ void final_dot_kernel(const float* __restrict__ z, const float* __restrict__ W3,
                                 const float* __restrict__ b3, float* __restrict__ out, int NB_) {
    int lane = threadIdx.x & 31;
    int row  = (int)((blockIdx.x * (long long)blockDim.x + threadIdx.x) >> 5);
    if (row >= NB_) return;
    float acc = 0.0f;
    for (int k = lane; k < 128; k += 32) acc += z[(size_t)row * 128 + k] * W3[k];
#pragma unroll
    for (int off = 16; off > 0; off >>= 1) acc += __shfl_xor(acc, off, 32);
    if (lane == 0) out[row] = acc + b3[0];
}

// ---------------------------------------------------------------------------
// Host orchestration
// ---------------------------------------------------------------------------
static void launch_gemm(const uint32_t* A, const uint32_t* Bt, const float* bias,
                        const float* addend, float* C, int M, int N, int Kpad, int relu,
                        hipStream_t s) {
    dim3 grid((unsigned)((M + 127) / 128), (unsigned)(N / 64));
    gemm_wmma_bf16<<<grid, dim3(256), 0, s>>>(A, Bt, bias, addend, C, M, N, Kpad, relu);
}
static void launch_cvt(const float* X, uint32_t* Y, int rows, int K, int Kpad, hipStream_t s) {
    int total = rows * (Kpad >> 1);
    cvt_pack_kernel<<<cdiv_l(total, 256), 256, 0, s>>>(X, Y, total, K, Kpad);
}
static void launch_cvtT(const float* W, uint32_t* Wt, int K, int N, int Kpad, hipStream_t s) {
    int total = N * (Kpad >> 1);
    cvtT_pack_kernel<<<cdiv_l(total, 256), 256, 0, s>>>(W, Wt, total, K, N, Kpad);
}
static void launch_zero(float* p, long long n, hipStream_t s) {
    zero_f32_kernel<<<cdiv_l(n, 256), 256, 0, s>>>(p, n);
}

extern "C" void kernel_launch(void* const* d_in, const int* in_sizes, int n_in,
                              void* d_out, int out_size, void* d_ws, size_t ws_size,
                              hipStream_t stream) {
    const float* x          = (const float*)d_in[0];
    const float* edge_attr  = (const float*)d_in[1];
    const int*   edge_index = (const int*)  d_in[2];
    const int*   solvent_ids= (const int*)  d_in[3];
    const int*   b_index    = (const int*)  d_in[4];
    const int*   batch_index= (const int*)  d_in[5];

    const int Nn = in_sizes[0] / 64;    // 20000
    const int Ee = in_sizes[1] / 16;    // 320000
    const int Gg = in_sizes[3];         // 512
    const int NB = in_sizes[4];         // 2048
    const int* srcI = edge_index;
    const int* dstI = edge_index + Ee;

    int pi = 7;
    const float* node_W = (const float*)d_in[pi++];
    const float* node_b = (const float*)d_in[pi++];
    const float* edge_W = (const float*)d_in[pi++];
    const float* edge_b = (const float*)d_in[pi++];
    const float* solvent_emb = (const float*)d_in[pi++];
    const float* vn_emb = (const float*)d_in[pi++];
    const float* mlp_W1 = (const float*)d_in[pi++];
    const float* mlp_b1 = (const float*)d_in[pi++];
    const float* mlp_W2 = (const float*)d_in[pi++];
    const float* mlp_b2 = (const float*)d_in[pi++];
    const float* mlp_W3 = (const float*)d_in[pi++];
    const float* mlp_b3 = (const float*)d_in[pi++];
    const float *Wq[3], *bq[3], *Wk[3], *bk[3], *Wv[3], *bv[3], *We[3], *Ws[3], *bs[3];
    const float *bng[3], *bnb[3], *vnW[3], *vnb[3], *vng[3], *vnbeta[3];
    for (int i = 0; i < 3; ++i) {
        Wq[i] = (const float*)d_in[pi++]; bq[i] = (const float*)d_in[pi++];
        Wk[i] = (const float*)d_in[pi++]; bk[i] = (const float*)d_in[pi++];
        Wv[i] = (const float*)d_in[pi++]; bv[i] = (const float*)d_in[pi++];
        We[i] = (const float*)d_in[pi++];
        Ws[i] = (const float*)d_in[pi++]; bs[i] = (const float*)d_in[pi++];
        bng[i] = (const float*)d_in[pi++]; bnb[i] = (const float*)d_in[pi++];
        vnW[i] = (const float*)d_in[pi++]; vnb[i] = (const float*)d_in[pi++];
        vng[i] = (const float*)d_in[pi++]; vnbeta[i] = (const float*)d_in[pi++];
    }

    char* wp = (char*)d_ws;
    auto alloc = [&](size_t bytes) -> void* {
        void* p = (void*)wp;
        wp += (bytes + 255) & ~(size_t)255;
        return p;
    };
    // bf16 A-operand buffers first (row-overshoot of the last 128-row tile
    // lands in the following allocations, which are always mapped)
    uint32_t* x_bf    = (uint32_t*)alloc((size_t)Nn * 32  * 4);  // [Nn][64/2]
    uint32_t* hh_bf   = (uint32_t*)alloc((size_t)Nn * 128 * 4);  // [Nn][256/2]
    uint32_t* edge_bf = (uint32_t*)alloc((size_t)Ee * 16  * 4);  // [Ee][32/2]
    uint32_t* e_bf    = (uint32_t*)alloc((size_t)Ee * 128 * 4);  // [Ee][256/2]
    uint32_t* pooled_bf = (uint32_t*)alloc((size_t)Gg * 128 * 4);
    uint32_t* comb_bf = (uint32_t*)alloc((size_t)NB * 160 * 4);  // [NB][320/2]
    uint32_t* z1_bf   = (uint32_t*)alloc((size_t)NB * 128 * 4);
    uint32_t* wT      = (uint32_t*)alloc((size_t)256 * 160 * 4); // transposed weight scratch
    float* e_proj = (float*)alloc((size_t)Ee * HIDDIM * 4);
    float* h      = (float*)alloc((size_t)Nn * HIDDIM * 4);
    float* Qb     = (float*)alloc((size_t)Nn * HIDDIM * 4);
    float* Kb     = (float*)alloc((size_t)Nn * HIDDIM * 4);
    float* Vb     = (float*)alloc((size_t)Nn * HIDDIM * 4);
    float* Ob     = (float*)alloc((size_t)Nn * HIDDIM * 4);
    float* convo  = (float*)alloc((size_t)Nn * HIDDIM * 4);
    float* alpha  = (float*)alloc((size_t)Ee * NHEADS * 4);
    unsigned* m_enc = (unsigned*)alloc((size_t)Nn * NHEADS * 4);
    float* ssum   = (float*)alloc((size_t)Nn * NHEADS * 4);
    float* pooled = (float*)alloc((size_t)Gg * HIDDIM * 4);
    float* vn     = (float*)alloc((size_t)Gg * HIDDIM * 4);
    float* vnz    = (float*)alloc((size_t)Gg * HIDDIM * 4);
    float* bsum   = (float*)alloc(HIDDIM * 4);
    float* bsq    = (float*)alloc(HIDDIM * 4);
    float* bmean  = (float*)alloc(HIDDIM * 4);
    float* brstd  = (float*)alloc(HIDDIM * 4);
    float* z1     = (float*)alloc((size_t)NB * 256 * 4);
    float* z2     = (float*)alloc((size_t)NB * 128 * 4);
    (void)alloc(262144);   // tail slack for tile overshoot safety
    (void)ws_size; (void)n_in;

    const long long nTot = (long long)Nn * HIDDIM;
    const long long eh   = (long long)Ee * NHEADS;

    // ---- input projections (pre-converted operands) ----
    launch_cvt(x, x_bf, Nn, 64, 64, stream);
    launch_cvt(edge_attr, edge_bf, Ee, 16, 32, stream);
    launch_cvtT(node_W, wT, 64, 256, 64, stream);
    launch_gemm(x_bf, wT, node_b, nullptr, h, Nn, HIDDIM, 64, 0, stream);
    launch_cvtT(edge_W, wT, 16, 256, 32, stream);
    launch_gemm(edge_bf, wT, edge_b, nullptr, e_proj, Ee, HIDDIM, 32, 0, stream);
    launch_cvt(e_proj, e_bf, Ee, HIDDIM, HIDDIM, stream);
    bcast_row_kernel<<<cdiv_l((long long)Gg * HIDDIM, 256), 256, 0, stream>>>(
        vn_emb, vn, (long long)Gg * HIDDIM);

    for (int L = 0; L < 3; ++L) {
        add_vn_cvt_kernel<<<cdiv_l((long long)Nn * 128, 256), 256, 0, stream>>>(
            h, vn, batch_index, hh_bf, Nn * 128);
        launch_cvtT(Wq[L], wT, HIDDIM, HIDDIM, HIDDIM, stream);
        launch_gemm(hh_bf, wT, bq[L], nullptr, Qb, Nn, HIDDIM, HIDDIM, 0, stream);
        launch_cvtT(Wk[L], wT, HIDDIM, HIDDIM, HIDDIM, stream);
        launch_gemm(hh_bf, wT, bk[L], nullptr, Kb, Nn, HIDDIM, HIDDIM, 0, stream);
        launch_cvtT(Wv[L], wT, HIDDIM, HIDDIM, HIDDIM, stream);
        launch_gemm(hh_bf, wT, bv[L], nullptr, Vb, Nn, HIDDIM, HIDDIM, 0, stream);
        launch_cvtT(We[L], wT, HIDDIM, HIDDIM, HIDDIM, stream);
        launch_gemm(e_bf, wT, nullptr, nullptr, e_proj, Ee, HIDDIM, HIDDIM, 0, stream); // ee
        launch_zero(Ob, nTot, stream);
        launch_zero(ssum, (long long)Nn * NHEADS, stream);
        zero_u32_kernel<<<cdiv_l((long long)Nn * NHEADS, 256), 256, 0, stream>>>(
            m_enc, (long long)Nn * NHEADS);
        attn_alpha_kernel<<<cdiv_l(eh, 256), 256, 0, stream>>>(
            Qb, Kb, e_proj, srcI, dstI, m_enc, alpha, eh);
        attn_expsum_kernel<<<cdiv_l(eh, 256), 256, 0, stream>>>(dstI, m_enc, alpha, ssum, eh);
        attn_scatter_kernel<<<cdiv_l((long long)Ee * 64, 256), 256, 0, stream>>>(
            Vb, e_proj, alpha, ssum, srcI, dstI, Ob, (long long)Ee * 64);
        launch_cvtT(Ws[L], wT, HIDDIM, HIDDIM, HIDDIM, stream);
        launch_gemm(hh_bf, wT, bs[L], Ob, convo, Nn, HIDDIM, HIDDIM, 0, stream);
        launch_zero(bsum, HIDDIM, stream);
        launch_zero(bsq, HIDDIM, stream);
        colsum_kernel<<<cdiv_l(Nn, 128), 256, 0, stream>>>(convo, bsum, bsq, Nn);
        bnstats_finalize_kernel<<<1, 256, 0, stream>>>(bsum, bsq, bmean, brstd, Nn);
        bn_relu_res_kernel<<<cdiv_l(nTot, 256), 256, 0, stream>>>(
            convo, bmean, brstd, bng[L], bnb[L], h, h, nTot);
        copy_f32_kernel<<<cdiv_l((long long)Gg * HIDDIM, 256), 256, 0, stream>>>(
            vn, pooled, (long long)Gg * HIDDIM);
        pool_scatter_kernel<<<cdiv_l((long long)Nn * 64, 256), 256, 0, stream>>>(
            h, batch_index, pooled, (long long)Nn * 64);
        launch_cvt(pooled, pooled_bf, Gg, HIDDIM, HIDDIM, stream);
        launch_cvtT(vnW[L], wT, HIDDIM, HIDDIM, HIDDIM, stream);
        launch_gemm(pooled_bf, wT, vnb[L], nullptr, vnz, Gg, HIDDIM, HIDDIM, 0, stream);
        launch_zero(bsum, HIDDIM, stream);
        launch_zero(bsq, HIDDIM, stream);
        colsum_kernel<<<cdiv_l(Gg, 128), 256, 0, stream>>>(vnz, bsum, bsq, Gg);
        bnstats_finalize_kernel<<<1, 256, 0, stream>>>(bsum, bsq, bmean, brstd, Gg);
        bn_relu_res_kernel<<<cdiv_l((long long)Gg * HIDDIM, 256), 256, 0, stream>>>(
            vnz, bmean, brstd, vng[L], vnbeta[L], nullptr, vn, (long long)Gg * HIDDIM);
    }

    gather_comb_cvt_kernel<<<cdiv_l((long long)NB * 160, 256), 256, 0, stream>>>(
        h, solvent_emb, solvent_ids, b_index, batch_index, comb_bf, NB * 160);
    launch_cvtT(mlp_W1, wT, 320, 256, 320, stream);
    launch_gemm(comb_bf, wT, mlp_b1, nullptr, z1, NB, 256, 320, 1, stream);
    launch_cvt(z1, z1_bf, NB, 256, 256, stream);
    launch_cvtT(mlp_W2, wT, 256, 128, 256, stream);
    launch_gemm(z1_bf, wT, mlp_b2, nullptr, z2, NB, 128, 256, 1, stream);
    final_dot_kernel<<<cdiv_l((long long)NB * 32, 256), 256, 0, stream>>>(
        z2, mlp_W3, mlp_b3, (float*)d_out, NB);
    (void)out_size;
}
